// Winograd2d_59030030516261
// MI455X (gfx1250) — matrix operand order repaired and checked by reading, not compiled, not run on hardware
//
#include <hip/hip_runtime.h>
#include <hip/hip_bf16.h>

// ---------------------------------------------------------------------------
// Winograd F(4x4) conv, 8x8 transform domain, via 64 batched bf16 WMMA GEMMs.
// N=8, IC=OC=128, HW=68 -> tH=tW=16, M = 8*16*16 = 2048 tiles.
// ws layout: wt bf16 [64][128][128] (2MB) | xt bf16 [64][2048][128] (32MB)
//            | yt f32 [64][2048][128] (64MB)   total ~98MB
// GEMM staging uses GLOBAL_LOAD_ASYNC_TO_LDS_B128 (ASYNCcnt) when the
// toolchain declares the builtin; otherwise global->VGPR->ds_store fallback.
// ---------------------------------------------------------------------------

typedef __attribute__((ext_vector_type(8)))  unsigned short v8us;
typedef __attribute__((ext_vector_type(16))) unsigned short v16us;
typedef __attribute__((ext_vector_type(16))) __bf16        v16bf;
typedef __attribute__((ext_vector_type(8)))  float         v8f;
typedef int v4i __attribute__((vector_size(16)));   // builtin's pointee type

#define NB   8
#define IC   128
#define OC   128
#define HW   68
#define TH   16
#define TW   16
#define MTOT (NB*TH*TW)   // 2048
#define NFREQ 64

#if defined(__has_builtin)
#if __has_builtin(__builtin_amdgcn_global_load_async_to_lds_b128) && \
    __has_builtin(__builtin_amdgcn_s_wait_asynccnt)
#define USE_ASYNC_LDS 1
#endif
#endif

__device__ __forceinline__ unsigned short f2bf(float f) {
    unsigned int u = __builtin_bit_cast(unsigned int, f);
    unsigned int r = u + 0x7FFFu + ((u >> 16) & 1u);   // RNE
    return (unsigned short)(r >> 16);
}

// ---------------------------------------------------------------------------
// Kernel 1: repack weights (OC,IC,8,8) f32 -> wt[ab][oc][ic] bf16
// ---------------------------------------------------------------------------
__global__ void wg_weight_repack(const float* __restrict__ w,
                                 unsigned short* __restrict__ wt) {
    int idx = blockIdx.x * 256 + threadIdx.x;           // 64*128*128
    if (idx >= NFREQ * OC * IC) return;
    int c  = idx & 127;
    int o  = (idx >> 7) & 127;
    int ab = idx >> 14;
    float v = w[(o * IC + c) * 64 + ab];
    wt[(ab * OC + o) * IC + c] = f2bf(v);
}

// ---------------------------------------------------------------------------
// Kernel 2: input transform  xt[ab][m][c] = (BT * d * BT^T)[a][b]  (bf16)
// ---------------------------------------------------------------------------
__global__ void wg_input_transform(const float* __restrict__ x,
                                   unsigned short* __restrict__ xt) {
    const float BT[8][8] = {
        {1.f, 0.f, -5.25f, 0.f, 5.25f, 0.f, -1.f, 0.f},
        {0.f, 1.f, 1.f, -4.25f, -4.25f, 1.f, 1.f, 0.f},
        {0.f, -1.f, 1.f, 4.25f, -4.25f, -1.f, 1.f, 0.f},
        {0.f, 0.5f, 0.25f, -2.5f, -1.25f, 2.f, 1.f, 0.f},
        {0.f, -0.5f, 0.25f, 2.5f, -1.25f, -2.f, 1.f, 0.f},
        {0.f, 2.f, 4.f, -2.5f, -5.f, 0.5f, 1.f, 0.f},
        {0.f, -2.f, 4.f, 2.5f, -5.f, -0.5f, 1.f, 0.f},
        {0.f, -1.f, 0.f, 5.25f, 0.f, -5.25f, 0.f, 1.f}};

    int idx = blockIdx.x * 256 + threadIdx.x;           // 8*16*16*128
    if (idx >= NB * TH * TW * IC) return;
    int c  = idx & 127;
    int t  = idx >> 7;
    int tw = t & 15;
    int th = (t >> 4) & 15;
    int n  = t >> 8;

    const float* xp = x + (((size_t)(n * IC + c) * HW + th * 4) * HW + tw * 4);
    float d[8][8];
#pragma unroll
    for (int i = 0; i < 8; ++i)
#pragma unroll
        for (int j = 0; j < 8; ++j)
            d[i][j] = xp[i * HW + j];

    float tm[8][8];                                     // tm = BT * d
#pragma unroll
    for (int a = 0; a < 8; ++a)
#pragma unroll
        for (int j = 0; j < 8; ++j) {
            float s = 0.f;
#pragma unroll
            for (int i = 0; i < 8; ++i) s += BT[a][i] * d[i][j];
            tm[a][j] = s;
        }

    int m = (n * TH + th) * TW + tw;
#pragma unroll
    for (int a = 0; a < 8; ++a)
#pragma unroll
        for (int b = 0; b < 8; ++b) {
            float s = 0.f;
#pragma unroll
            for (int j = 0; j < 8; ++j) s += tm[a][j] * BT[b][j];
            xt[((size_t)(a * 8 + b) * MTOT + m) * IC + c] = f2bf(s);
        }
}

// ---------------------------------------------------------------------------
// Kernel 3: batched GEMM  yt[ab][m][o] = sum_c xt[ab][m][c] * wt[ab][o][c]
// block = 256 threads (8 waves), tile 128(M) x 128(N), K stepped by 32,
// double-buffered LDS, v_wmma_f32_16x16x32_bf16.
// ---------------------------------------------------------------------------
#define LDA 40   // padded row length in halfs (80B rows: conflict-free, 16B aligned)

__global__ void wg_gemm(const unsigned short* __restrict__ xt,
                        const unsigned short* __restrict__ wt,
                        float* __restrict__ yt) {
    __shared__ __align__(16) unsigned short As[2][128 * LDA];
    __shared__ __align__(16) unsigned short Bs[2][128 * LDA];

    const int tid   = threadIdx.x;
    const int ab    = blockIdx.y;
    const int mBase = blockIdx.x * 128;

    const unsigned short* Ap = xt + (size_t)ab * MTOT * IC + (size_t)mBase * IC;
    const unsigned short* Bp = wt + (size_t)ab * OC * IC;
    float* Cp = yt + (size_t)ab * MTOT * OC + (size_t)mBase * OC;

    const int lane  = tid & 31;
    const int wid   = tid >> 5;
    const int mg    = wid & 3;   // wave M group: 4 groups of 32 rows
    const int ng    = wid >> 2;  // wave N group: 2 groups of 64 cols
    const int half0 = lane & 15;
    const int sel   = lane >> 4; // lane-half selects K sub-range

    v8f acc[2][4] = {};

    // ---- tile loader: 128x32 halfs each for A and B, 2 x 16B chunks/thread
    auto loadTile = [&](int ks, int buf) {
#pragma unroll
        for (int p = 0; p < 2; ++p) {
            int q    = tid + p * 256;        // chunk id 0..511
            int row  = q >> 2;               // 0..127
            int off8 = (q & 3) * 8;          // 0,8,16,24 halfs
            const unsigned short* ga = Ap + (size_t)row * IC + ks * 32 + off8;
            const unsigned short* gb = Bp + (size_t)row * IC + ks * 32 + off8;
            unsigned short* la = &As[buf][row * LDA + off8];
            unsigned short* lb = &Bs[buf][row * LDA + off8];
#ifdef USE_ASYNC_LDS
            // CDNA5 async copy: GLOBAL_LOAD_ASYNC_TO_LDS_B128 (ASYNCcnt),
            // no VGPR staging. Args: (global src, LDS dst, imm offset, cpol).
            __builtin_amdgcn_global_load_async_to_lds_b128(
                (v4i*)ga, (v4i*)la, 0, 0);
            __builtin_amdgcn_global_load_async_to_lds_b128(
                (v4i*)gb, (v4i*)lb, 0, 0);
#else
            v8us av = *(const v8us*)ga;
            v8us bv = *(const v8us*)gb;
            *(v8us*)la = av;
            *(v8us*)lb = bv;
#endif
        }
    };
    auto waitTiles = [&]() {
#ifdef USE_ASYNC_LDS
        __builtin_amdgcn_s_wait_asynccnt(0);   // LDS writes visible pre-barrier
#endif
    };

    loadTile(0, 0);
    waitTiles();
    __syncthreads();

#pragma unroll
    for (int ks = 0; ks < IC / 32; ++ks) {
        int cur = ks & 1;
        if (ks + 1 < IC / 32) loadTile(ks + 1, cur ^ 1);

        // A fragments (ISA 16-bit A layout: row = M%16 per lane-half,
        // K halfs {kb..kb+7} U {kb+16..kb+23}, kb = sel*8)
        v16bf af[2];
#pragma unroll
        for (int i = 0; i < 2; ++i) {
            int base = (mg * 32 + i * 16 + half0) * LDA + sel * 8;
            v8us a0 = *(const v8us*)&As[cur][base];
            v8us a1 = *(const v8us*)&As[cur][base + 16];
            v16us cat = __builtin_shufflevector(a0, a1,
                0, 1, 2, 3, 4, 5, 6, 7, 8, 9, 10, 11, 12, 13, 14, 15);
            af[i] = __builtin_bit_cast(v16bf, cat);
        }
        // B fragments (column n = N%16 per lane, K halfs {kb..kb+7} U {kb+16..kb+23}, kb = sel*8)
        v16bf bf[4];
#pragma unroll
        for (int j = 0; j < 4; ++j) {
            int base = (ng * 64 + j * 16 + half0) * LDA + sel * 8;
            v8us b0 = *(const v8us*)&Bs[cur][base];
            v8us b1 = *(const v8us*)&Bs[cur][base + 16];
            v16us cat = __builtin_shufflevector(b0, b1,
                0, 1, 2, 3, 4, 5, 6, 7, 8, 9, 10, 11, 12, 13, 14, 15);
            bf[j] = __builtin_bit_cast(v16bf, cat);
        }

#pragma unroll
        for (int i = 0; i < 2; ++i)
#pragma unroll
            for (int j = 0; j < 4; ++j)
                acc[i][j] = __builtin_amdgcn_wmma_f32_16x16x32_bf16(
                    false, af[i], false, bf[j], (short)0, acc[i][j],
                    false, false);

        waitTiles();
        __syncthreads();
    }

    // C/D layout: elem e -> M = e + sel*8 ; N = lane%16
#pragma unroll
    for (int i = 0; i < 2; ++i)
#pragma unroll
        for (int j = 0; j < 4; ++j) {
            int col = ng * 64 + j * 16 + half0;
#pragma unroll
            for (int e = 0; e < 8; ++e) {
                int row = mg * 32 + i * 16 + sel * 8 + e;
                Cp[(size_t)row * OC + col] = acc[i][j][e];
            }
        }
}

// ---------------------------------------------------------------------------
// Kernel 4: output transform  out = AT * Y * AT^T + bias
// ---------------------------------------------------------------------------
__global__ void wg_output_transform(const float* __restrict__ yt,
                                    const float* __restrict__ bias,
                                    float* __restrict__ out) {
    const float AT[4][8] = {
        {1.f, 1.f, 1.f, 1.f, 1.f, 8.f, 8.f, 0.f},
        {0.f, 1.f, -1.f, 2.f, -2.f, 4.f, -4.f, 0.f},
        {0.f, 1.f, 1.f, 4.f, 4.f, 2.f, 2.f, 0.f},
        {0.f, 1.f, -1.f, 8.f, -8.f, 1.f, -1.f, 1.f}};

    int idx = blockIdx.x * 256 + threadIdx.x;           // 8*16*16*128
    if (idx >= NB * TH * TW * OC) return;
    int o  = idx & 127;
    int t  = idx >> 7;
    int tw = t & 15;
    int th = (t >> 4) & 15;
    int n  = t >> 8;
    int m  = (n * TH + th) * TW + tw;

    float Y[8][8];
#pragma unroll
    for (int a = 0; a < 8; ++a)
#pragma unroll
        for (int b = 0; b < 8; ++b)
            Y[a][b] = yt[((size_t)(a * 8 + b) * MTOT + m) * OC + o];

    float u[4][8];
#pragma unroll
    for (int p = 0; p < 4; ++p)
#pragma unroll
        for (int b = 0; b < 8; ++b) {
            float s = 0.f;
#pragma unroll
            for (int a = 0; a < 8; ++a) s += AT[p][a] * Y[a][b];
            u[p][b] = s;
        }

    float bv = bias[o];
    float* op = out + (((size_t)(n * OC + o) * 64 + th * 4) * 64 + tw * 4);
#pragma unroll
    for (int p = 0; p < 4; ++p)
#pragma unroll
        for (int q = 0; q < 4; ++q) {
            float s = 0.f;
#pragma unroll
            for (int b = 0; b < 8; ++b) s += u[p][b] * AT[q][b];
            op[p * 64 + q] = s + bv;
        }
}

// ---------------------------------------------------------------------------
extern "C" void kernel_launch(void* const* d_in, const int* in_sizes, int n_in,
                              void* d_out, int out_size, void* d_ws, size_t ws_size,
                              hipStream_t stream) {
    const float* x    = (const float*)d_in[0];  // (8,128,68,68)
    const float* w    = (const float*)d_in[1];  // (128,128,8,8)
    const float* bias = (const float*)d_in[2];  // (128,)
    float* out = (float*)d_out;                 // (8,128,64,64)

    char* ws = (char*)d_ws;
    const size_t WT_BYTES = (size_t)NFREQ * OC * IC * sizeof(unsigned short);   // 2 MB
    const size_t XT_BYTES = (size_t)NFREQ * MTOT * IC * sizeof(unsigned short); // 32 MB
    unsigned short* wt = (unsigned short*)ws;
    unsigned short* xt = (unsigned short*)(ws + WT_BYTES);
    float*          yt = (float*)(ws + WT_BYTES + XT_BYTES);                    // 64 MB

    wg_weight_repack<<<(NFREQ * OC * IC + 255) / 256, 256, 0, stream>>>(w, wt);
    wg_input_transform<<<(NB * TH * TW * IC + 255) / 256, 256, 0, stream>>>(x, xt);
    wg_gemm<<<dim3(MTOT / 128, NFREQ), 256, 0, stream>>>(xt, wt, yt);
    wg_output_transform<<<(NB * TH * TW * OC + 255) / 256, 256, 0, stream>>>(yt, bias, out);
}
